// CRF_89747636617709
// MI455X (gfx1250) — matrix operand order repaired and checked by reading, not compiled, not run on hardware
//
#include <hip/hip_runtime.h>

typedef __attribute__((ext_vector_type(16))) _Float16 v16h;
typedef __attribute__((ext_vector_type(8)))  float    v8f;
typedef __attribute__((ext_vector_type(4)))  float    v4f;
typedef __attribute__((ext_vector_type(4)))  int      v4i;

#define Bn 1024
#define Tn 512
#define Ln 64
#define START_TAG 62
#define STOP_TAG  63
#define AST 68   // alpha row stride: 68*4 = 272 bytes -> every row 16B aligned,
                 // bank(tag,row) = (4*row + tag) % 64 -> column walks conflict-free
#define EST 68   // emission tile row stride (same property)
#define LOG2E 1.4426950408889634f
#define LN2   0.6931471805599453f
#define NEG2  (-14427.0f)   // -10000 * log2(e), the "minus infinity" in log2 domain

#if __has_builtin(__builtin_amdgcn_global_load_async_to_lds_b128) && \
    __has_builtin(__builtin_amdgcn_s_wait_asynccnt)
#define HAVE_ASYNC 1
#else
#define HAVE_ASYNC 0
#endif

static __device__ __forceinline__ v4f vmax4(v4f a, v4f b) {
    v4f r;
    r.x = fmaxf(a.x, b.x); r.y = fmaxf(a.y, b.y);
    r.z = fmaxf(a.z, b.z); r.w = fmaxf(a.w, b.w);
    return r;
}
static __device__ __forceinline__ float vget(v4f v, int i) {
    return i == 0 ? v.x : i == 1 ? v.y : i == 2 ? v.z : v.w;
}
static __device__ __forceinline__ void copy16_to_lds(const float* g, float* l) {
#if HAVE_ASYNC
    float* gnc = const_cast<float*>(g);
    __builtin_amdgcn_global_load_async_to_lds_b128(
        (__attribute__((address_space(1))) v4i*)gnc,
        (__attribute__((address_space(3))) v4i*)l, 0, 0);
#else
    *(v4f*)l = *(const v4f*)g;
#endif
}

// ---------------------------------------------------------------------------
// CRF forward scan in log2 domain. One workgroup (4 waves) per 16-seq tile.
// beta = alpha*log2e;  per step:
//   beta'[b,j] = emit[b,t,j]*log2e + m_b + log2( sum_i exp2(beta[b,i]-m_b) * E[i,j] )
// with E[i,j] = exp2(trans[j,i]*log2e) constant in WMMA B-fragments.
// ---------------------------------------------------------------------------
__global__ __launch_bounds__(128) void crf_forward_logz(
    const float* __restrict__ emissions,    // [B,T,L]
    const float* __restrict__ transitions,  // [L,L] trans[next, prev]
    const int*   __restrict__ lengths,      // [B]
    float*       __restrict__ logz)         // [B]
{
    __shared__ float alpha[2][16 * AST];
    __shared__ float etile[2][16 * EST];
    __shared__ float mrow[16];

    const int tid  = threadIdx.x;
    const int lane = tid & 31;
    const int wave = tid >> 5;
    const int hlf  = lane >> 4;
    const int col  = lane & 15;
    const int n0   = wave * 16;
    const int b0   = blockIdx.x * 16;

    // ---- init beta0 ----
    for (int i = tid; i < 16 * AST; i += 128) {
        int tag = i % AST;
        alpha[0][i] = (tag == START_TAG) ? 0.0f : NEG2;
    }

    // ---- constant B fragments: B[k,n] = exp2(trans[n,k]*log2e) ----
    v16h bfrag[2];
    {
        const int n = n0 + col;
#pragma unroll
        for (int c = 0; c < 2; ++c)
#pragma unroll
            for (int e = 0; e < 16; ++e) {
                int k = 32 * c + 16 * (e >> 3) + 8 * hlf + (e & 7);
                bfrag[c][e] =
                    (_Float16)__builtin_amdgcn_exp2f(transitions[n * Ln + k] * LOG2E);
            }
    }

    int lenr[8];
#pragma unroll
    for (int r = 0; r < 8; ++r) lenr[r] = lengths[b0 + r + hlf * 8];

    // ---- stage emission tile for t=0 (2 x b128 per thread) ----
    const int erow = tid >> 3;          // 0..15
    const int eq   = tid & 7;           // 0..7
    {
        const float* g = &emissions[((size_t)(b0 + erow) * Tn) * Ln + eq * 4];
        copy16_to_lds(g,      &etile[0][erow * EST + eq * 4]);
        copy16_to_lds(g + 32, &etile[0][erow * EST + eq * 4 + 32]);
    }
    __syncthreads();

    int cur = 0, eb = 0;
    const int arow = col;   // A-matrix row (batch within tile) for this lane
    for (int t = 0; t < Tn; ++t) {
        const float* aC = alpha[cur];
        const float* rp = aC + arow * AST;
        const float* pm = rp + 8 * hlf;        // this half's K blocks {0,16,32,48}+8h
        const float* po = rp + 8 * (1 - hlf);  // the other 32 values (max only)

        v4f mq[8], oq[8];
#pragma unroll
        for (int b_ = 0; b_ < 4; ++b_) {
            mq[2 * b_]     = *(const v4f*)(pm + 16 * b_);
            mq[2 * b_ + 1] = *(const v4f*)(pm + 16 * b_ + 4);
            oq[2 * b_]     = *(const v4f*)(po + 16 * b_);
            oq[2 * b_ + 1] = *(const v4f*)(po + 16 * b_ + 4);
        }
        // ---- tree max over the 64-tag row (depth ~6) ----
        v4f x0 = vmax4(vmax4(mq[0], mq[1]), vmax4(mq[2], mq[3]));
        v4f x1 = vmax4(vmax4(mq[4], mq[5]), vmax4(mq[6], mq[7]));
        v4f x2 = vmax4(vmax4(oq[0], oq[1]), vmax4(oq[2], oq[3]));
        v4f x3 = vmax4(vmax4(oq[4], oq[5]), vmax4(oq[6], oq[7]));
        v4f xm = vmax4(vmax4(x0, x1), vmax4(x2, x3));
        float m = fmaxf(fmaxf(xm.x, xm.y), fmaxf(xm.z, xm.w));
        if (wave == 0 && lane < 16) mrow[lane] = m;

        // ---- A fragments: exp2(beta - m) from the register copy ----
        v16h af[2];
#pragma unroll
        for (int c = 0; c < 2; ++c)
#pragma unroll
            for (int e = 0; e < 16; ++e) {
                int blk = 2 * c + (e >= 8 ? 1 : 0);
                int idx = 2 * blk + ((e & 7) >> 2);
                float v = vget(mq[idx], e & 3);
                af[c][e] = (_Float16)__builtin_amdgcn_exp2f(v - m);
            }

        // ---- pipeline emissions: issue t+1, drain t ----
        if (t + 1 < Tn) {
            const float* g =
                &emissions[((size_t)(b0 + erow) * Tn + (t + 1)) * Ln + eq * 4];
            copy16_to_lds(g,      &etile[eb ^ 1][erow * EST + eq * 4]);
            copy16_to_lds(g + 32, &etile[eb ^ 1][erow * EST + eq * 4 + 32]);
#if HAVE_ASYNC
            __builtin_amdgcn_s_wait_asynccnt(2);
#endif
        } else {
#if HAVE_ASYNC
            __builtin_amdgcn_s_wait_asynccnt(0);
#endif
        }

        __syncthreads();  // mrow + etile[eb] published; alpha[cur] reads done

        // ---- V = U x E : two chained v_wmma_f32_16x16x32_f16 ----
        v8f acc = {0.f, 0.f, 0.f, 0.f, 0.f, 0.f, 0.f, 0.f};
        acc = __builtin_amdgcn_wmma_f32_16x16x32_f16(
            false, af[0], false, bfrag[0], (short)0, acc, false, false);
        acc = __builtin_amdgcn_wmma_f32_16x16x32_f16(
            false, af[1], false, bfrag[1], (short)0, acc, false, false);

        // ---- beta' = emit*log2e + m + log2(V), masked by t < length ----
        float* aN = alpha[cur ^ 1];
        const float* eT = etile[eb];
#pragma unroll
        for (int r = 0; r < 8; ++r) {
            int mr = r + hlf * 8;
            float lg = __builtin_amdgcn_logf(acc[r]);   // raw v_log_f32 (log2)
            float nv = fmaf(eT[mr * EST + n0 + col], LOG2E, mrow[mr] + lg);
            float ov = aC[mr * AST + n0 + col];
            aN[mr * AST + n0 + col] = (t < lenr[r]) ? nv : ov;
        }
        __syncthreads();
        cur ^= 1;
        eb ^= 1;
    }

    // ---- log_z[b] = ln2 * log2sumexp2_j(beta[b,j] + trans[STOP,j]*log2e) ----
    if (wave == 0 && lane < 16) {
        const float* aF = alpha[cur] + lane * AST;
        float m2 = -3.4e38f;
#pragma unroll 8
        for (int j = 0; j < Ln; ++j)
            m2 = fmaxf(m2, aF[j] + transitions[STOP_TAG * Ln + j] * LOG2E);
        float s = 0.0f;
#pragma unroll 8
        for (int j = 0; j < Ln; ++j)
            s += __builtin_amdgcn_exp2f(
                aF[j] + transitions[STOP_TAG * Ln + j] * LOG2E - m2);
        logz[b0 + lane] = LN2 * (m2 + __builtin_amdgcn_logf(s));
    }
}

// ---------------------------------------------------------------------------
// Gold path score + final NLL. One thread per sequence.
// ---------------------------------------------------------------------------
__global__ __launch_bounds__(256) void crf_gold_nll(
    const float* __restrict__ emissions,
    const float* __restrict__ transitions,
    const int*   __restrict__ tags,
    const int*   __restrict__ lengths,
    const float* __restrict__ logz,
    float*       __restrict__ out)
{
    int b = blockIdx.x * 256 + threadIdx.x;
    if (b >= Bn) return;
    const int len = lengths[b];
    int prev = START_TAG;
    float acc = 0.0f;
    for (int t = 0; t < len; ++t) {
        int tag = tags[(size_t)b * Tn + t];
        acc += transitions[tag * Ln + prev] +
               emissions[((size_t)b * Tn + t) * Ln + tag];
        prev = tag;
    }
    acc += transitions[STOP_TAG * Ln + prev];
    out[b] = logz[b] - acc;
}

extern "C" void kernel_launch(void* const* d_in, const int* in_sizes, int n_in,
                              void* d_out, int out_size, void* d_ws, size_t ws_size,
                              hipStream_t stream) {
    const float* emissions   = (const float*)d_in[0];
    const float* transitions = (const float*)d_in[1];
    const int*   tags        = (const int*)d_in[2];
    const int*   lengths     = (const int*)d_in[3];
    float* out  = (float*)d_out;
    float* logz = (float*)d_ws;   // B floats of scratch

    crf_forward_logz<<<Bn / 16, 128, 0, stream>>>(emissions, transitions, lengths, logz);
    crf_gold_nll<<<Bn / 256, 256, 0, stream>>>(emissions, transitions, tags, lengths,
                                               logz, out);
}
